// DLSMNLayer_76570676953638
// MI455X (gfx1250) — compile-verified, matrix operand order checked by reading, not run on hardware
//
#include <hip/hip_runtime.h>
#include <math.h>

// ---------------------------------------------------------------------------
// DLSMN layer forward for MI455X (gfx1250): f16 WMMA GEMMs + flash attention.
// ---------------------------------------------------------------------------

typedef __attribute__((ext_vector_type(16))) _Float16 v16h;
typedef __attribute__((ext_vector_type(8)))  float    v8f;

#define DEV __device__ __forceinline__

static constexpr int Bx = 8, Sx = 1024, Dx = 1024, Hx = 8;
static constexpr int DCx = 128, TSx = 6144;
static constexpr size_t MROWS = (size_t)Bx * Sx;          // 8192

DEV v8f wmma_f16(v16h a, v16h b, v8f c) {
  // D = A(16x32 f16) * B(32x16 f16) + C(16x16 f32)
  return __builtin_amdgcn_wmma_f32_16x16x32_f16(false, a, false, b, (short)0, c,
                                                false, false);
}

// A-fragment / row-major-B fragment: lane holds one row r=lane&15; half-wave
// selects K-phase (lanes 0-15: K=kh+0..7 & 16+kh.., kh=0; lanes 16-31: kh=8).
DEV v16h ld_frag(const _Float16* p, int stride, int lane) {
  const int r  = lane & 15;
  const int kh = (lane >> 4) << 3;
  const _Float16* q = p + r * stride + kh;
  v16h f;
#pragma unroll
  for (int j = 0; j < 8; ++j) { f[j] = q[j]; f[8 + j] = q[16 + j]; }
  return f;
}

// B-fragment from a row-major K x N tile (column n per lane, strided rows).
DEV v16h ld_frag_col(const _Float16* p, int stride, int lane) {
  const int n  = lane & 15;
  const int kh = (lane >> 4) << 3;
  v16h f;
#pragma unroll
  for (int j = 0; j < 8; ++j) {
    f[j]     = p[(kh + j) * stride + n];
    f[8 + j] = p[(16 + kh + j) * stride + n];
  }
  return f;
}

// ---------------------------------------------------------------------------
// Generic GEMM: C[M,N] = act(A[M,K] @ W[N,K]^T + bias [+ C]) * rowscale
// Block tile 256x128, 8 waves in a 4x2 grid, 64x64 per wave (16 WMMA / K-step
// against 8 LDS fragment loads -> matrix pipe stays fed). K-step 32.
// Staging: all 12 b128 global loads issued into registers BEFORE the barrier
// so their latency overlaps the previous tile's WMMA tail; one wait covers
// the whole batch instead of 12 serialized round trips.
// Requires M%256==0, N%128==0, K%32==0.
// act: 0=none, 1=gelu(exact), 2=sigmoid
// ---------------------------------------------------------------------------
__global__ __launch_bounds__(256) void gemm_wmma_kernel(
    const float* __restrict__ A, int lda,
    const float* __restrict__ W, int ldw,
    const float* __restrict__ bias,
    const float* __restrict__ rowscale,
    float* __restrict__ C, int ldc,
    int K, int act, int accum)
{
  __shared__ _Float16 As[256][40];
  __shared__ _Float16 Bs[128][40];
  const int tid  = threadIdx.x;
  const int lane = tid & 31;
  const int wave = tid >> 5;
  const int wm = wave >> 1;       // 0..3 (M 64-blocks)
  const int wn = wave & 1;        // 0..1 (N 64-blocks)
  const long bm = (long)blockIdx.x * 256;
  const long bn = (long)blockIdx.y * 128;
  const int ar = tid >> 3;        // 0..31
  const int ac = (tid & 7) * 4;

  v8f acc[4][4] = {};

  for (int k0 = 0; k0 < K; k0 += 32) {
    // Batched global loads for this K slice (registers only; no LDS hazard,
    // so they are issued before the barrier and pipeline behind one wait).
    float4 areg[8], breg[4];
#pragma unroll
    for (int p = 0; p < 8; ++p)
      areg[p] = *(const float4*)(A + (size_t)(bm + p * 32 + ar) * lda + (k0 + ac));
#pragma unroll
    for (int p = 0; p < 4; ++p)
      breg[p] = *(const float4*)(W + (size_t)(bn + p * 32 + ar) * ldw + (k0 + ac));
    if (k0 + 32 < K)
      __builtin_prefetch(A + (size_t)(bm + ar) * lda + (k0 + 32 + ac), 0, 1);

    __syncthreads();   // previous tile fully consumed
#pragma unroll
    for (int p = 0; p < 8; ++p) {
      As[p * 32 + ar][ac + 0] = (_Float16)areg[p].x;
      As[p * 32 + ar][ac + 1] = (_Float16)areg[p].y;
      As[p * 32 + ar][ac + 2] = (_Float16)areg[p].z;
      As[p * 32 + ar][ac + 3] = (_Float16)areg[p].w;
    }
#pragma unroll
    for (int p = 0; p < 4; ++p) {
      Bs[p * 32 + ar][ac + 0] = (_Float16)breg[p].x;
      Bs[p * 32 + ar][ac + 1] = (_Float16)breg[p].y;
      Bs[p * 32 + ar][ac + 2] = (_Float16)breg[p].z;
      Bs[p * 32 + ar][ac + 3] = (_Float16)breg[p].w;
    }
    __syncthreads();   // tile ready

    v16h bf[4];
#pragma unroll
    for (int j = 0; j < 4; ++j)
      bf[j] = ld_frag(&Bs[wn * 64 + j * 16][0], 40, lane);
#pragma unroll
    for (int i = 0; i < 4; ++i) {
      v16h af = ld_frag(&As[wm * 64 + i * 16][0], 40, lane);
#pragma unroll
      for (int j = 0; j < 4; ++j)
        acc[i][j] = wmma_f16(af, bf[j], acc[i][j]);
    }
  }

  const int n  = lane & 15;
  const int mb = (lane >> 4) << 3;
#pragma unroll
  for (int ti = 0; ti < 4; ++ti)
#pragma unroll
    for (int tj = 0; tj < 4; ++tj) {
      long col = bn + wn * 64 + tj * 16 + n;
      float bv = bias ? bias[col] : 0.f;
#pragma unroll
      for (int g = 0; g < 8; ++g) {
        long row = bm + wm * 64 + ti * 16 + mb + g;
        float v = acc[ti][tj][g] + bv;
        size_t idx = (size_t)row * ldc + col;
        if (accum) v += C[idx];
        if (act == 1)      v = 0.5f * v * (1.f + erff(v * 0.70710678118654752f));
        else if (act == 2) v = 1.f / (1.f + __expf(-v));
        if (rowscale) v *= rowscale[row];
        C[idx] = v;
      }
    }
}

// ---------------------------------------------------------------------------
// Flash attention core: each wave owns 16 queries x 128 head-dim; block = 4
// waves (64 queries). Streams keys in chunks of 32 through f16 LDS tiles.
// CACHE mode adds dead-slot bias (-1e30) and MoE slot bias for t in [384,512).
// ---------------------------------------------------------------------------
template <bool CACHE>
DEV void flash_core(const float* __restrict__ Qb, int ldq,
                    const float* __restrict__ Kb, int ldk,
                    const float* __restrict__ Vb, int ldv,
                    float* __restrict__ Ob, int ldo,
                    int Tlen,
                    const float* __restrict__ deadb,   // [Tlen]
                    const float* __restrict__ probsb)  // [64][128]
{
  __shared__ _Float16 Kt[32][136];
  __shared__ _Float16 Vt[32][136];
  __shared__ _Float16 Pt[4][16][40];
  const int tid  = threadIdx.x;   // 128 threads
  const int lane = tid & 31;
  const int wave = tid >> 5;      // query sub-tile 0..3
  const int n    = lane & 15;
  const int mb   = (lane >> 4) << 3;
  const float scale = 0.08838834764831845f;  // 1/sqrt(128)

  // Q fragments kept in VGPRs (A layout, 4 K-steps of 32), b128 gathers.
  const float* Qw = Qb + (size_t)(wave * 16) * ldq;
  const int r  = lane & 15;
  const int kh = (lane >> 4) << 3;
  v16h qf[4];
#pragma unroll
  for (int ks = 0; ks < 4; ++ks) {
    const float* src = Qw + (size_t)r * ldq + ks * 32 + kh;
    float4 q0 = *(const float4*)(src);
    float4 q1 = *(const float4*)(src + 4);
    float4 q2 = *(const float4*)(src + 16);
    float4 q3 = *(const float4*)(src + 20);
    v16h f;
    f[0]  = (_Float16)q0.x; f[1]  = (_Float16)q0.y; f[2]  = (_Float16)q0.z; f[3]  = (_Float16)q0.w;
    f[4]  = (_Float16)q1.x; f[5]  = (_Float16)q1.y; f[6]  = (_Float16)q1.z; f[7]  = (_Float16)q1.w;
    f[8]  = (_Float16)q2.x; f[9]  = (_Float16)q2.y; f[10] = (_Float16)q2.z; f[11] = (_Float16)q2.w;
    f[12] = (_Float16)q3.x; f[13] = (_Float16)q3.y; f[14] = (_Float16)q3.z; f[15] = (_Float16)q3.w;
    qf[ks] = f;
  }

  float m_r[8], l_r[8];
  v8f accO[8] = {};
#pragma unroll
  for (int g = 0; g < 8; ++g) { m_r[g] = -3.0e38f; l_r[g] = 0.f; }

  const int srow = tid >> 2;           // 0..31 staging row
  const int scol = (tid & 3) * 32;     // 0..96 staging col base

  for (int t0 = 0; t0 < Tlen; t0 += 32) {
    // Batched K/V loads into registers before the barrier (8 x b128 each;
    // each thread covers one row x 32-col span).
    float4 kreg[8], vreg[8];
#pragma unroll
    for (int p = 0; p < 8; ++p) {
      kreg[p] = *(const float4*)(Kb + (size_t)(t0 + srow) * ldk + scol + p * 4);
      vreg[p] = *(const float4*)(Vb + (size_t)(t0 + srow) * ldv + scol + p * 4);
    }
    if (t0 + 32 < Tlen) {
      __builtin_prefetch(Kb + (size_t)(t0 + 32 + srow) * ldk + scol, 0, 1);
      __builtin_prefetch(Vb + (size_t)(t0 + 32 + srow) * ldv + scol, 0, 1);
    }
    __syncthreads();   // previous tile fully consumed
#pragma unroll
    for (int p = 0; p < 8; ++p) {
      Kt[srow][scol + p * 4 + 0] = (_Float16)kreg[p].x;
      Kt[srow][scol + p * 4 + 1] = (_Float16)kreg[p].y;
      Kt[srow][scol + p * 4 + 2] = (_Float16)kreg[p].z;
      Kt[srow][scol + p * 4 + 3] = (_Float16)kreg[p].w;
      Vt[srow][scol + p * 4 + 0] = (_Float16)vreg[p].x;
      Vt[srow][scol + p * 4 + 1] = (_Float16)vreg[p].y;
      Vt[srow][scol + p * 4 + 2] = (_Float16)vreg[p].z;
      Vt[srow][scol + p * 4 + 3] = (_Float16)vreg[p].w;
    }
    __syncthreads();   // tile ready

    // S = Q @ K^T for two 16-key column chunks.
    v8f s0 = {}, s1 = {};
#pragma unroll
    for (int ks2 = 0; ks2 < 4; ++ks2) {
      v16h b0 = ld_frag(&Kt[0][ks2 * 32],  136, lane);
      v16h b1 = ld_frag(&Kt[16][ks2 * 32], 136, lane);
      s0 = wmma_f16(qf[ks2], b0, s0);
      s1 = wmma_f16(qf[ks2], b1, s1);
    }

    if (CACHE) {
      const int ta = t0 + n, tb = t0 + 16 + n;
      const float d0 = deadb[ta], d1 = deadb[tb];
      const bool in0 = (ta >= 384 && ta < 512);
      const bool in1 = (tb >= 384 && tb < 512);
#pragma unroll
      for (int g = 0; g < 8; ++g) {
        const size_t qrow = (size_t)(wave * 16 + mb + g) * 128;
        float e0 = s0[g] * scale + d0;
        float e1 = s1[g] * scale + d1;
        if (in0) e0 += 5.f * probsb[qrow + (ta - 384)];
        if (in1) e1 += 5.f * probsb[qrow + (tb - 384)];
        s0[g] = e0; s1[g] = e1;
      }
    } else {
#pragma unroll
      for (int g = 0; g < 8; ++g) { s0[g] *= scale; s1[g] *= scale; }
    }

    // Online softmax: per-row (16 lanes of a half-wave hold the 16 columns).
#pragma unroll
    for (int g = 0; g < 8; ++g) {
      float mx = fmaxf(s0[g], s1[g]);
#pragma unroll
      for (int m = 1; m < 16; m <<= 1) mx = fmaxf(mx, __shfl_xor(mx, m, 32));
      float mn = fmaxf(m_r[g], mx);
      float p0 = __expf(s0[g] - mn);
      float p1 = __expf(s1[g] - mn);
      s0[g] = p0; s1[g] = p1;
      float rs = p0 + p1;
#pragma unroll
      for (int m = 1; m < 16; m <<= 1) rs += __shfl_xor(rs, m, 32);
      float alpha = __expf(m_r[g] - mn);
      m_r[g] = mn;
      l_r[g] = l_r[g] * alpha + rs;
#pragma unroll
      for (int h = 0; h < 8; ++h) accO[h][g] *= alpha;
    }

    // C-layout P -> A-fragment via wave-private LDS tile (DS ops are in-order
    // within a wave; compiler barrier stops reordering).
#pragma unroll
    for (int g = 0; g < 8; ++g) {
      Pt[wave][mb + g][n]      = (_Float16)s0[g];
      Pt[wave][mb + g][16 + n] = (_Float16)s1[g];
    }
    __asm__ volatile("" ::: "memory");
    v16h pf = ld_frag(&Pt[wave][0][0], 40, lane);
#pragma unroll
    for (int h = 0; h < 8; ++h) {
      v16h bv = ld_frag_col(&Vt[0][h * 16], 136, lane);
      accO[h] = wmma_f16(pf, bv, accO[h]);
    }
  }

  float* Ow = Ob + (size_t)(wave * 16) * ldo;
#pragma unroll
  for (int g = 0; g < 8; ++g) {
    float inv = 1.f / l_r[g];
#pragma unroll
    for (int h = 0; h < 8; ++h)
      Ow[(size_t)(mb + g) * ldo + h * 16 + n] = accO[h][g] * inv;
  }
}

__global__ __launch_bounds__(128) void flash_cache_kernel(
    const float* __restrict__ Q, const float* __restrict__ K,
    const float* __restrict__ V, float* __restrict__ O,
    const float* __restrict__ deadb, const float* __restrict__ probs)
{
  const int b  = blockIdx.x;               // 0..7
  const int qt = blockIdx.y;               // 0..15
  const size_t qrow = (size_t)b * Sx + qt * 64;
  flash_core<true>(Q + qrow * DCx, DCx,
                   K + (size_t)b * TSx * DCx, DCx,
                   V + (size_t)b * TSx * DCx, DCx,
                   O + qrow * 160, 160,
                   TSx,
                   deadb + (size_t)b * TSx,
                   probs + qrow * 128);
}

__global__ __launch_bounds__(128) void flash_mha_kernel(
    const float* __restrict__ Q, const float* __restrict__ K,
    const float* __restrict__ V, float* __restrict__ O)
{
  const int bh = blockIdx.x;               // 0..63
  const int b = bh >> 3, h = bh & 7;
  const size_t qrow = (size_t)b * Sx + blockIdx.y * 64;
  const size_t hoff = (size_t)h * 128;
  flash_core<false>(Q + qrow * Dx + hoff, Dx,
                    K + (size_t)b * Sx * Dx + hoff, Dx,
                    V + (size_t)b * Sx * Dx + hoff, Dx,
                    O + qrow * Dx + hoff, Dx,
                    Sx, nullptr, nullptr);
}

// ---------------------------------------------------------------------------
// Fused residual add + LayerNorm: out = LN(a + b) * g + beta   (one row/block)
// ---------------------------------------------------------------------------
__global__ __launch_bounds__(256) void add_ln_kernel(
    const float* __restrict__ a, const float* __restrict__ bsrc,
    const float* __restrict__ g, const float* __restrict__ beta,
    float* __restrict__ out, int Dn)
{
  __shared__ float red[16];
  const size_t row = blockIdx.x;
  const float* pa = a + row * Dn;
  const float* pb = bsrc + row * Dn;
  float s = 0.f, ss = 0.f;
  for (int i = threadIdx.x; i < Dn; i += 256) {
    float v = pa[i] + pb[i];
    s += v; ss += v * v;
  }
#pragma unroll
  for (int m = 1; m < 32; m <<= 1) { s += __shfl_xor(s, m, 32); ss += __shfl_xor(ss, m, 32); }
  const int wave = threadIdx.x >> 5, lane = threadIdx.x & 31;
  if (lane == 0) { red[wave] = s; red[8 + wave] = ss; }
  __syncthreads();
  float ts = 0.f, tss = 0.f;
#pragma unroll
  for (int w = 0; w < 8; ++w) { ts += red[w]; tss += red[8 + w]; }
  const float mean = ts / Dn;
  const float var  = tss / Dn - mean * mean;
  const float rstd = rsqrtf(var + 1e-5f);
  for (int i = threadIdx.x; i < Dn; i += 256) {
    float v = pa[i] + pb[i];
    out[row * Dn + i] = (v - mean) * rstd * g[i] + beta[i];
  }
}

// xf = g*xm + (1-g)*x   (g read from and xf written to the same buffer)
__global__ void gate_fuse_kernel(float* __restrict__ gxf,
                                 const float* __restrict__ xm,
                                 const float* __restrict__ x, size_t N)
{
  size_t i = (size_t)blockIdx.x * blockDim.x + threadIdx.x;
  if (i < N) {
    float gv = gxf[i];
    gxf[i] = gv * xm[i] + (1.f - gv) * x[i];
  }
}

// ---------------------------------------------------------------------------
// Prep kernels
// ---------------------------------------------------------------------------
// cwi[b,t,:] = [cache | layer_ids[t] | ages/100 | zero-pad to 192]
__global__ void prep_cwi_kernel(const float* __restrict__ cache,
                                const float* __restrict__ layer_ids,
                                const float* __restrict__ ages,
                                float* __restrict__ cwi)
{
  size_t i = (size_t)blockIdx.x * blockDim.x + threadIdx.x;
  if (i >= (size_t)Bx * TSx * 192) return;
  int col = (int)(i % 192);
  size_t rowg = i / 192;
  int t = (int)(rowg % TSx);
  float v;
  if (col < 128)      v = cache[rowg * 128 + col];
  else if (col < 160) v = layer_ids[(size_t)t * 32 + (col - 128)];
  else if (col == 160) v = ages[rowg] * 0.01f;
  else                 v = 0.f;
  cwi[i] = v;
}

__global__ void prep_dead_kernel(const unsigned char* __restrict__ mask,
                                 const int* __restrict__ wc,
                                 float* __restrict__ deadb, int N)
{
  int i = blockIdx.x * blockDim.x + threadIdx.x;
  if (i < N) deadb[i] = (mask[i] != 0 || wc[i] == 0) ? -1e30f : 0.f;
}

// raw[:,128:160] = layer_embed
__global__ void prep_rawtail_kernel(const float* __restrict__ le,
                                    float* __restrict__ raw)
{
  int i = blockIdx.x * blockDim.x + threadIdx.x;
  if (i >= (int)MROWS * 32) return;
  int row = i >> 5, j = i & 31;
  raw[(size_t)row * 160 + 128 + j] = le[j];
}

// pad [128,161] weight to [128,192] with zeros
__global__ void prep_wpad_kernel(const float* __restrict__ w,
                                 float* __restrict__ wp)
{
  int i = blockIdx.x * blockDim.x + threadIdx.x;
  if (i >= 128 * 192) return;
  int rw = i / 192, c = i % 192;
  wp[i] = (c < 161) ? w[rw * 161 + c] : 0.f;
}

// ---------------------------------------------------------------------------
// Host-side orchestration
// ---------------------------------------------------------------------------
static void gemm(const float* A, int lda, const float* W, int ldw,
                 const float* bias, const float* rowscale,
                 float* C, int ldc, int M, int N, int K, int act, int accum,
                 hipStream_t st)
{
  dim3 grid(M / 256, N / 128);
  gemm_wmma_kernel<<<grid, 256, 0, st>>>(A, lda, W, ldw, bias, rowscale,
                                         C, ldc, K, act, accum);
}

extern "C" void kernel_launch(void* const* d_in, const int* in_sizes, int n_in,
                              void* d_out, int out_size, void* d_ws, size_t ws_size,
                              hipStream_t stream)
{
  (void)in_sizes; (void)n_in; (void)out_size; (void)ws_size;
  const float* x           = (const float*)d_in[0];
  const float* cache       = (const float*)d_in[1];
  const float* layer_ids   = (const float*)d_in[2];
  const float* slot_ages   = (const float*)d_in[3];
  const float* read_gate   = (const float*)d_in[4];
  const float* slot_probs  = (const float*)d_in[5];
  const float* layer_embed = (const float*)d_in[6];
  const float* cq_w = (const float*)d_in[7];   const float* cq_b = (const float*)d_in[8];
  const float* ck_w = (const float*)d_in[9];   const float* ck_b = (const float*)d_in[10];
  const float* cv_w = (const float*)d_in[11];  const float* cv_b = (const float*)d_in[12];
  const float* dec_w = (const float*)d_in[13]; const float* dec_b = (const float*)d_in[14];
  const float* cwin  = (const float*)d_in[15]; const float* cbin  = (const float*)d_in[16];
  const float* cwout = (const float*)d_in[17]; const float* cbout = (const float*)d_in[18];
  const float* ncg = (const float*)d_in[19];   const float* ncb = (const float*)d_in[20];
  const float* gate_w = (const float*)d_in[21]; const float* gate_b = (const float*)d_in[22];
  const float* swin  = (const float*)d_in[23]; const float* sbin  = (const float*)d_in[24];
  const float* swout = (const float*)d_in[25]; const float* sbout = (const float*)d_in[26];
  const float* n1g = (const float*)d_in[27];   const float* n1b = (const float*)d_in[28];
  const float* fw1 = (const float*)d_in[29];   const float* fb1 = (const float*)d_in[30];
  const float* fw2 = (const float*)d_in[31];   const float* fb2 = (const float*)d_in[32];
  const float* n2g = (const float*)d_in[33];   const float* n2b = (const float*)d_in[34];
  const unsigned char* cache_mask = (const unsigned char*)d_in[35];
  const int*   write_counts       = (const int*)d_in[36];
  float* out = (float*)d_out;

  // Workspace arena (floats). RegionA is reused three times.
  float* WS = (float*)d_ws;
  float* RA = WS;                         // 33,554,432 floats
  float* RB = WS + 33554432;              //  8,388,608
  float* RC = WS + 41943040;              //  8,388,608
  float* RD = WS + 50331648;              //  8,388,608
  // stage-1 sub-layout inside RegionA:
  float* cwi  = RA;                       // 49152 x 192
  float* kbuf = RA + 9437184;             // 49152 x 128
  float* vbuf = RA + 15728640;            // 49152 x 128
  float* qbuf = RA + 22020096;            //  8192 x 128
  float* dead = RA + 23068672;            // 49152
  float* raw  = RA + 23117824;            //  8192 x 160
  // padded cache-key/value weights live in RegionD during stage 1:
  float* ckw_pad = RD;                    // 128 x 192
  float* cvw_pad = RD + 24576;            // 128 x 192
  // MHA QKV slices inside RegionA (stages 3 and 5):
  float* QA = RA;
  float* KA = RA + 8388608;
  float* VA = RA + 16777216;

  // ---- prep --------------------------------------------------------------
  {
    size_t n = (size_t)Bx * TSx * 192;
    prep_cwi_kernel<<<(unsigned)((n + 255) / 256), 256, 0, stream>>>(
        cache, layer_ids, slot_ages, cwi);
    prep_dead_kernel<<<(Bx * TSx + 255) / 256, 256, 0, stream>>>(
        cache_mask, write_counts, dead, Bx * TSx);
    prep_rawtail_kernel<<<((int)MROWS * 32 + 255) / 256, 256, 0, stream>>>(
        layer_embed, raw);
    prep_wpad_kernel<<<(128 * 192 + 255) / 256, 256, 0, stream>>>(ck_w, ckw_pad);
    prep_wpad_kernel<<<(128 * 192 + 255) / 256, 256, 0, stream>>>(cv_w, cvw_pad);
  }

  // ---- READ: cache attention ---------------------------------------------
  gemm(cwi, 192, ckw_pad, 192, ck_b, nullptr, kbuf, 128, Bx * TSx, 128, 192, 0, 0, stream);
  gemm(cwi, 192, cvw_pad, 192, cv_b, nullptr, vbuf, 128, Bx * TSx, 128, 192, 0, 0, stream);
  gemm(x, Dx, cq_w, Dx, cq_b, nullptr, qbuf, 128, (int)MROWS, 128, Dx, 0, 0, stream);
  {
    dim3 grid(Bx, Sx / 64);
    flash_cache_kernel<<<grid, 128, 0, stream>>>(qbuf, kbuf, vbuf, raw, dead, slot_probs);
  }
  // ctx = (raw @ dec_w^T + dec_b) * read_gate  -> RB
  gemm(raw, 160, dec_w, 160, dec_b, read_gate, RB, Dx, (int)MROWS, Dx, 160, 0, 0, stream);

  // ---- cross attention (q from x, kv from ctx) ---------------------------
  gemm(x,  Dx, cwin,                        Dx, cbin,        nullptr, QA, Dx, (int)MROWS, Dx, Dx, 0, 0, stream);
  gemm(RB, Dx, cwin + (size_t)1024 * 1024,  Dx, cbin + 1024, nullptr, KA, Dx, (int)MROWS, Dx, Dx, 0, 0, stream);
  gemm(RB, Dx, cwin + (size_t)2048 * 1024,  Dx, cbin + 2048, nullptr, VA, Dx, (int)MROWS, Dx, Dx, 0, 0, stream);
  {
    dim3 grid(Bx * Hx, Sx / 64);
    flash_mha_kernel<<<grid, 128, 0, stream>>>(QA, KA, VA, RC);
  }
  gemm(RC, Dx, cwout, Dx, cbout, nullptr, RB, Dx, (int)MROWS, Dx, Dx, 0, 0, stream); // cross -> RB
  add_ln_kernel<<<(unsigned)MROWS, 256, 0, stream>>>(x, RB, ncg, ncb, RC, Dx);        // xm -> RC

  // ---- gated fusion: g = sigmoid(x@W1^T + xm@W2^T + b); xf = g*xm+(1-g)*x -
  gemm(x,  Dx, gate_w,        2048, gate_b, nullptr, RB, Dx, (int)MROWS, Dx, Dx, 0, 0, stream);
  gemm(RC, Dx, gate_w + 1024, 2048, nullptr, nullptr, RB, Dx, (int)MROWS, Dx, Dx, 2, 1, stream);
  gate_fuse_kernel<<<(unsigned)((MROWS * Dx + 255) / 256), 256, 0, stream>>>(
      RB, RC, x, MROWS * Dx);                                                         // xf -> RB

  // ---- self attention -----------------------------------------------------
  gemm(RB, Dx, swin,                        Dx, sbin,        nullptr, QA, Dx, (int)MROWS, Dx, Dx, 0, 0, stream);
  gemm(RB, Dx, swin + (size_t)1024 * 1024,  Dx, sbin + 1024, nullptr, KA, Dx, (int)MROWS, Dx, Dx, 0, 0, stream);
  gemm(RB, Dx, swin + (size_t)2048 * 1024,  Dx, sbin + 2048, nullptr, VA, Dx, (int)MROWS, Dx, Dx, 0, 0, stream);
  {
    dim3 grid(Bx * Hx, Sx / 64);
    flash_mha_kernel<<<grid, 128, 0, stream>>>(QA, KA, VA, RC);
  }
  gemm(RC, Dx, swout, Dx, sbout, nullptr, RD, Dx, (int)MROWS, Dx, Dx, 0, 0, stream);  // sa -> RD
  add_ln_kernel<<<(unsigned)MROWS, 256, 0, stream>>>(RB, RD, n1g, n1b, RC, Dx);       // xf2 -> RC

  // ---- FFN ----------------------------------------------------------------
  gemm(RC, Dx, fw1, Dx,   fb1, nullptr, RA, 4096, (int)MROWS, 4096, Dx,   1, 0, stream); // h -> RA
  gemm(RA, 4096, fw2, 4096, fb2, nullptr, RD, Dx, (int)MROWS, Dx,   4096, 0, 0, stream); // -> RD
  add_ln_kernel<<<(unsigned)MROWS, 256, 0, stream>>>(RC, RD, n2g, n2b, out, Dx);      // y
}
